// EquivariantBlock_75574244540703
// MI455X (gfx1250) — matrix operand order repaired and checked by reading, not compiled, not run on hardware
//
#include <hip/hip_runtime.h>
#include <hip/hip_bf16.h>
#include <math.h>

// ---------------------------------------------------------------------------
// EquivariantBlock for MI455X (gfx1250).
// Compute-bound (1.37 TFLOP vs <1 GB traffic) -> bf16 WMMA w/ f32 accumulate.
//   pass 1: LayerNorm (f32 in, bf16 out)
//   pass 2: transpose+convert W1, W2 to bf16 (B-fragment-friendly layout)
//   pass 3: GEMM1 (rolled A) + bias + fast tanh-GELU -> h (bf16)
//   pass 4: GEMM2 + bias + residual -> out (f32)
// Wave tile 64x64 (16 WMMA / K-step), block 128x256, branch-free GELU.
// ---------------------------------------------------------------------------

typedef __bf16 bf16_t;
typedef __attribute__((ext_vector_type(16))) __bf16 v16bf;
typedef __attribute__((ext_vector_type(8)))  __bf16 v8bf;
typedef __attribute__((ext_vector_type(4)))  __bf16 v4bf;
typedef __attribute__((ext_vector_type(8)))  float  v8f;

static constexpr int kBatch = 8192;
static constexpr int kWidth = 1024;        // width
static constexpr int kMid   = 4096;        // mid
static constexpr int kK1    = 4096;        // 4*width
static constexpr int kRows  = kBatch * 4;  // 32768 independent GEMM rows

// ---------------------------------------------------------------------------
// Pass 1: LayerNorm over last dim (1024). One 256-thread block per row.
// ---------------------------------------------------------------------------
__global__ void __launch_bounds__(256)
ln_kernel(const float* __restrict__ x, const float* __restrict__ sc,
          const float* __restrict__ of, bf16_t* __restrict__ yb)
{
    const int row = blockIdx.x;
    const int tid = threadIdx.x;
    const float4 xv = ((const float4*)(x + (size_t)row * kWidth))[tid];

    float s  = xv.x + xv.y + xv.z + xv.w;
    float ss = xv.x * xv.x + xv.y * xv.y + xv.z * xv.z + xv.w * xv.w;
    #pragma unroll
    for (int off = 16; off > 0; off >>= 1) {
        s  += __shfl_down(s,  off, 32);
        ss += __shfl_down(ss, off, 32);
    }
    __shared__ float lsum[8], lsumsq[8];
    const int wave = tid >> 5, lane = tid & 31;
    if (lane == 0) { lsum[wave] = s; lsumsq[wave] = ss; }
    __syncthreads();
    if (tid == 0) {
        float S = 0.f, SS = 0.f;
        #pragma unroll
        for (int i = 0; i < 8; ++i) { S += lsum[i]; SS += lsumsq[i]; }
        const float mu  = S * (1.0f / kWidth);
        const float var = SS * (1.0f / kWidth) - mu * mu;
        lsum[0]   = mu;
        lsumsq[0] = rsqrtf(var + 1e-5f);
    }
    __syncthreads();
    const float mu = lsum[0], rs = lsumsq[0];

    v4bf o;
    const float xs[4] = { xv.x, xv.y, xv.z, xv.w };
    #pragma unroll
    for (int i = 0; i < 4; ++i) {
        const int c = tid * 4 + i;
        o[i] = (bf16_t)((xs[i] - mu) * rs * sc[c] + of[c]);
    }
    ((v4bf*)(yb + (size_t)row * kWidth))[tid] = o;
}

// ---------------------------------------------------------------------------
// Pass 2: out[c][r] = bf16(in[r][c]).  in: R x C f32, out: C x R bf16.
// ---------------------------------------------------------------------------
__global__ void __launch_bounds__(256)
transpose_to_bf16_kernel(const float* __restrict__ in, bf16_t* __restrict__ out,
                         int R, int C)
{
    __shared__ float tile[32][33];
    const int tx = threadIdx.x, ty = threadIdx.y;
    const int c0 = blockIdx.x * 32;
    const int r0 = blockIdx.y * 32;
    #pragma unroll
    for (int i = 0; i < 4; ++i)
        tile[ty + i * 8][tx] = in[(size_t)(r0 + ty + i * 8) * C + (c0 + tx)];
    __syncthreads();
    #pragma unroll
    for (int i = 0; i < 4; ++i)
        out[(size_t)(c0 + ty + i * 8) * R + (r0 + tx)] = (bf16_t)tile[tx][ty + i * 8];
}

// ---------------------------------------------------------------------------
// WMMA fragment helpers (ISA 7.12.2 layouts, wave32).
// A 16x32 bf16: lane m=lane&15, half=lane>>4; v0..v3 = K[h*8 .. h*8+7],
//               v4..v7 = K[16+h*8 .. 16+h*8+7]   -> two b128 loads.
// B 32x16 bf16 (rows of W^T): lane n=lane&15; v0..v3 = K[h*8 .. h*8+7],
//               v4..v7 = K[16+h*8 .. 16+h*8+7]   -> two b128 loads.
// C/D f32 16x16: VGPR r <-> M = r + 8*half, N = lane&15.
// ---------------------------------------------------------------------------
__device__ __forceinline__ v16bf load_frag_2x8(const bf16_t* p0, const bf16_t* p1)
{
    union { v16bf v; v8bf h[2]; } u;
    u.h[0] = *(const v8bf*)p0;
    u.h[1] = *(const v8bf*)p1;
    return u.v;
}

// Branch-free tanh-GELU: gelu(x) = x * sigmoid(2*k0*(x + k1 x^3)).
// exp->inf => rcp->0 (x<<0), exp->0 => rcp->1 (x>>0): correct limits, no branches.
__device__ __forceinline__ float gelu_fast(float x)
{
    const float k0 = 0.7978845608028654f;  // sqrt(2/pi)
    const float k1 = 0.044715f;
    const float z = k0 * (x + k1 * x * x * x);
    return x * __builtin_amdgcn_rcpf(1.0f + __expf(-2.0f * z));
}

// ---------------------------------------------------------------------------
// Pass 3: h = gelu(rolls(y) @ W1 + b1), bf16 out.
// 8 waves/block; wave tile 64(M) x 64(N); block tile 128 x 256.
// Outer loop over the 4 roll segments (j): A row = y[b*4 + (j+k)&3], so the
// per-lane A row pointer is constant inside a segment (pure pointer marching).
// ---------------------------------------------------------------------------
__global__ void __launch_bounds__(256)
gemm1_gelu_kernel(const bf16_t* __restrict__ yb, const bf16_t* __restrict__ w1t,
                  const float* __restrict__ b1, bf16_t* __restrict__ hbuf)
{
    const int lane = threadIdx.x & 31;
    const int wave = threadIdx.x >> 5;
    const int wm   = wave & 1;    // 2 waves along M
    const int wn   = wave >> 1;   // 4 waves along N
    const int half = lane >> 4;
    const int lm   = lane & 15;

    const int Mblk = blockIdx.x * 128 + wm * 64;  // wave rows [Mblk, Mblk+64)
    const int Nblk = blockIdx.y * 256 + wn * 64;  // wave cols [Nblk, Nblk+64)

    v8f acc[4][4];
    #pragma unroll
    for (int t = 0; t < 4; ++t)
        #pragma unroll
        for (int n = 0; n < 4; ++n)
            acc[t][n] = (v8f){};

    int rbase[4], kr[4];
    #pragma unroll
    for (int t = 0; t < 4; ++t) {
        const int gr = Mblk + t * 16 + lm;  // global A row (b*4 + k)
        rbase[t] = gr & ~3;
        kr[t]    = gr & 3;
    }
    const bf16_t* wrow[4];
    #pragma unroll
    for (int n = 0; n < 4; ++n)
        wrow[n] = w1t + (size_t)(Nblk + n * 16 + lm) * kK1;

    for (int j = 0; j < 4; ++j) {                 // roll segment: cols [j*1024, ..)
        const bf16_t* arow[4];
        #pragma unroll
        for (int t = 0; t < 4; ++t)
            arow[t] = yb + ((size_t)(rbase[t] + ((j + kr[t]) & 3)) << 10);

        for (int kk = 0; kk < 1024; kk += 32) {   // chunks never cross the segment
            v16bf a[4], b[4];
            #pragma unroll
            for (int t = 0; t < 4; ++t) {
                const bf16_t* p = arow[t] + kk + half * 8;
                a[t] = load_frag_2x8(p, p + 16);
            }
            #pragma unroll
            for (int n = 0; n < 4; ++n) {
                const bf16_t* p = wrow[n] + (j << 10) + kk + half * 8;
                b[n] = load_frag_2x8(p, p + 16);
            }
            #pragma unroll
            for (int t = 0; t < 4; ++t)
                #pragma unroll
                for (int n = 0; n < 4; ++n)
                    acc[t][n] = __builtin_amdgcn_wmma_f32_16x16x32_bf16(
                        false, a[t], false, b[n], (short)0, acc[t][n], false, false);
        }
    }

    #pragma unroll
    for (int t = 0; t < 4; ++t) {
        #pragma unroll
        for (int n = 0; n < 4; ++n) {
            const int col  = Nblk + n * 16 + lm;
            const float bs = b1[col];
            #pragma unroll
            for (int r = 0; r < 8; ++r) {
                const int rowi = Mblk + t * 16 + r + half * 8;
                hbuf[(size_t)rowi * kMid + col] = (bf16_t)gelu_fast(acc[t][n][r] + bs);
            }
        }
    }
}

// ---------------------------------------------------------------------------
// Pass 4: out = x + h @ W2 + b2, f32 out. Same tiling, A = h (no roll).
// ---------------------------------------------------------------------------
__global__ void __launch_bounds__(256)
gemm2_resid_kernel(const bf16_t* __restrict__ hbuf, const bf16_t* __restrict__ w2t,
                   const float* __restrict__ b2, const float* __restrict__ x,
                   float* __restrict__ out)
{
    const int lane = threadIdx.x & 31;
    const int wave = threadIdx.x >> 5;
    const int wm   = wave & 1;
    const int wn   = wave >> 1;
    const int half = lane >> 4;
    const int lm   = lane & 15;

    const int Mblk = blockIdx.x * 128 + wm * 64;
    const int Nblk = blockIdx.y * 256 + wn * 64;

    v8f acc[4][4];
    #pragma unroll
    for (int t = 0; t < 4; ++t)
        #pragma unroll
        for (int n = 0; n < 4; ++n)
            acc[t][n] = (v8f){};

    const bf16_t* arow[4];
    #pragma unroll
    for (int t = 0; t < 4; ++t)
        arow[t] = hbuf + (size_t)(Mblk + t * 16 + lm) * kMid;
    const bf16_t* wrow[4];
    #pragma unroll
    for (int n = 0; n < 4; ++n)
        wrow[n] = w2t + (size_t)(Nblk + n * 16 + lm) * kMid;

    for (int kk = 0; kk < kMid; kk += 32) {
        v16bf a[4], b[4];
        #pragma unroll
        for (int t = 0; t < 4; ++t) {
            const bf16_t* p = arow[t] + kk + half * 8;
            a[t] = load_frag_2x8(p, p + 16);
        }
        #pragma unroll
        for (int n = 0; n < 4; ++n) {
            const bf16_t* p = wrow[n] + kk + half * 8;
            b[n] = load_frag_2x8(p, p + 16);
        }
        #pragma unroll
        for (int t = 0; t < 4; ++t)
            #pragma unroll
            for (int n = 0; n < 4; ++n)
                acc[t][n] = __builtin_amdgcn_wmma_f32_16x16x32_bf16(
                    false, a[t], false, b[n], (short)0, acc[t][n], false, false);
    }

    #pragma unroll
    for (int t = 0; t < 4; ++t) {
        #pragma unroll
        for (int n = 0; n < 4; ++n) {
            const int col  = Nblk + n * 16 + lm;
            const float bs = b2[col];
            #pragma unroll
            for (int r = 0; r < 8; ++r) {
                const int rowi = Mblk + t * 16 + r + half * 8;
                const size_t idx = (size_t)rowi * kWidth + col;
                out[idx] = x[idx] + acc[t][n][r] + bs;
            }
        }
    }
}

// ---------------------------------------------------------------------------
// Launch. Inputs: x, ln_scale, ln_offset, W1, b1, W2, b2 (all f32).
// Workspace (360 MB): y bf16 @0, W1^T bf16 @64MB, W2^T bf16 @96MB, h bf16 @104MB.
// ---------------------------------------------------------------------------
extern "C" void kernel_launch(void* const* d_in, const int* in_sizes, int n_in,
                              void* d_out, int out_size, void* d_ws, size_t ws_size,
                              hipStream_t stream)
{
    const float* x  = (const float*)d_in[0];
    const float* sc = (const float*)d_in[1];
    const float* of = (const float*)d_in[2];
    const float* W1 = (const float*)d_in[3];
    const float* b1 = (const float*)d_in[4];
    const float* W2 = (const float*)d_in[5];
    const float* b2 = (const float*)d_in[6];
    float* out = (float*)d_out;

    char* ws = (char*)d_ws;
    bf16_t* yb   = (bf16_t*)(ws);
    bf16_t* w1t  = (bf16_t*)(ws + (size_t)64  * 1024 * 1024);
    bf16_t* w2t  = (bf16_t*)(ws + (size_t)96  * 1024 * 1024);
    bf16_t* hbuf = (bf16_t*)(ws + (size_t)104 * 1024 * 1024);

    // 1) LayerNorm: 32768 rows
    ln_kernel<<<kRows, 256, 0, stream>>>(x, sc, of, yb);
    // 2) Weight transpose+convert
    transpose_to_bf16_kernel<<<dim3(kMid / 32, kK1 / 32), dim3(32, 8), 0, stream>>>(W1, w1t, kK1, kMid);
    transpose_to_bf16_kernel<<<dim3(kWidth / 32, kMid / 32), dim3(32, 8), 0, stream>>>(W2, w2t, kMid, kWidth);
    // 3) GEMM1 + GELU: block tile 128x256 -> grid (256, 16)
    gemm1_gelu_kernel<<<dim3(kRows / 128, kMid / 256), 256, 0, stream>>>(yb, w1t, b1, hbuf);
    // 4) GEMM2 + residual: grid (256, 4)
    gemm2_resid_kernel<<<dim3(kRows / 128, kWidth / 256), 256, 0, stream>>>(hbuf, w2t, b2, x, out);
}
